// MLA_39986145525983
// MI455X (gfx1250) — compile-verified, matrix operand order checked by reading, not run on hardware
//
#include <hip/hip_runtime.h>

#define S_LEN 4096
#define DIM   2048
#define NH    16
#define DNOPE 128
#define DROPE 64
#define DHEAD 192
#define DV    128
#define QLORA 1024
#define KVLORA 512
#define KDIM  576
#define FA_SCALE 0.07216878364870323f   // 192^-0.5
#define EPS 1e-6f

typedef __attribute__((ext_vector_type(16))) __bf16 bf16x16;
typedef __attribute__((ext_vector_type(8)))  float  f32x8;
typedef unsigned short u16;
typedef unsigned int   u32;
typedef long long      i64;

// ---- gfx1250 async global->LDS (ASYNCcnt) path, with sync fallback --------
#if defined(__AMDGCN__) && __has_builtin(__builtin_amdgcn_global_load_async_to_lds_b128)
#define HAVE_ASYNC_LDS 1
typedef int v4i __attribute__((vector_size(16)));
typedef __attribute__((address_space(1))) v4i as1_v4i;
typedef __attribute__((address_space(3))) v4i as3_v4i;
__device__ __forceinline__ void async_copy_b128(const u16* g, u16* l) {
  __builtin_amdgcn_global_load_async_to_lds_b128(
      (as1_v4i*)(void*)g, (as3_v4i*)(void*)l, 0, 0);
}
__device__ __forceinline__ void wait_async() {
#if __has_builtin(__builtin_amdgcn_s_wait_asynccnt)
  __builtin_amdgcn_s_wait_asynccnt(0);
#else
  asm volatile("s_wait_asynccnt 0" ::: "memory");
#endif
}
#else
#define HAVE_ASYNC_LDS 0
#endif

__device__ __forceinline__ u16 f2bf(float f) {
  u32 u = __float_as_uint(f);
  u32 r = u + 0x7fffu + ((u >> 16) & 1u);   // round to nearest even
  return (u16)(r >> 16);
}

// Load a 16-bit A/B WMMA fragment (16x32) given this lane's row pointer at the
// k-chunk base.  half = lane/16 selects K {0..7,16..23} vs {8..15,24..31}.
__device__ __forceinline__ bf16x16 load_frag(const u16* p, int half) {
  union { uint4 u[2]; bf16x16 v; } t;
  t.u[0] = *reinterpret_cast<const uint4*>(p + half * 8);
  t.u[1] = *reinterpret_cast<const uint4*>(p + 16 + half * 8);
  return t.v;
}

// ---------------------------------------------------------------------------
// Generic bf16 WMMA GEMM:  C[M,N] = A[M,K] * B[N,K]^T   (fp32 accumulate)
// 128x128 block tile, BK=32, 8 waves (2 in M x 4 in N), wave tile 64x32.
// Double-buffered LDS; tile k+1 streamed with async global->LDS DMA while
// WMMAs consume tile k, closed with s_wait_asynccnt + barrier.
// ---------------------------------------------------------------------------
#define GBM 128
#define GBN 128
#define GBK 32
#define GLD 40   // LDS row stride (bf16 elems), 80B = multiple of 16B

template<bool OUT_BF16>
__global__ __launch_bounds__(256)
void gemm_bf16(const u16* __restrict__ A, const u16* __restrict__ B,
               void* __restrict__ Cv, int M, int N, int K,
               int lda, int ldb, int ldc,
               i64 aBatch, i64 bBatch, i64 cBatch)
{
  __shared__ u16 As[2][GBM * GLD];
  __shared__ u16 Bs[2][GBN * GLD];
  const int z = blockIdx.z;
  A += (i64)z * aBatch;
  B += (i64)z * bBatch;
  const int m0 = blockIdx.y * GBM;
  const int n0 = blockIdx.x * GBN;
  const int tid  = threadIdx.x;
  const int wave = tid >> 5, lane = tid & 31;
  const int half = lane >> 4, lrow = lane & 15;
  const int wm = (wave & 1) * 64;   // 2 waves across M
  const int wn = (wave >> 1) * 32;  // 4 waves across N

  f32x8 acc[4][2] = {};

  const int r0 = tid >> 2;             // 0..63
  const int cc = (tid & 3) * 8;        // 0,8,16,24

  auto load_tile = [&](int k0, int buf) {
    #pragma unroll
    for (int rr0 = 0; rr0 < GBM; rr0 += 64) {
      const int rr = rr0 + r0;
      const int gm = m0 + rr;
#if HAVE_ASYNC_LDS
      // OOB rows skipped: garbage LDS only feeds masked-out C elements.
      if (gm < M) async_copy_b128(A + (i64)gm * lda + k0 + cc, &As[buf][rr * GLD + cc]);
#else
      uint4 v = make_uint4(0u, 0u, 0u, 0u);
      if (gm < M) v = *reinterpret_cast<const uint4*>(A + (i64)gm * lda + k0 + cc);
      *reinterpret_cast<uint4*>(&As[buf][rr * GLD + cc]) = v;
#endif
    }
    #pragma unroll
    for (int rr0 = 0; rr0 < GBN; rr0 += 64) {
      const int rr = rr0 + r0;
      const int gn = n0 + rr;
#if HAVE_ASYNC_LDS
      if (gn < N) async_copy_b128(B + (i64)gn * ldb + k0 + cc, &Bs[buf][rr * GLD + cc]);
#else
      uint4 v = make_uint4(0u, 0u, 0u, 0u);
      if (gn < N) v = *reinterpret_cast<const uint4*>(B + (i64)gn * ldb + k0 + cc);
      *reinterpret_cast<uint4*>(&Bs[buf][rr * GLD + cc]) = v;
#endif
    }
  };

  load_tile(0, 0);
#if HAVE_ASYNC_LDS
  wait_async();
#endif
  __syncthreads();

  const int nk = K / GBK;
  for (int ki = 0; ki < nk; ++ki) {
    const int buf = ki & 1;
    if (ki + 1 < nk) load_tile((ki + 1) * GBK, buf ^ 1);   // overlap with WMMAs

    bf16x16 bfr[2];
    #pragma unroll
    for (int tn = 0; tn < 2; ++tn)
      bfr[tn] = load_frag(&Bs[buf][(wn + tn * 16 + lrow) * GLD], half);
    #pragma unroll
    for (int tm = 0; tm < 4; ++tm) {
      bf16x16 afr = load_frag(&As[buf][(wm + tm * 16 + lrow) * GLD], half);
      #pragma unroll
      for (int tn = 0; tn < 2; ++tn)
        acc[tm][tn] = __builtin_amdgcn_wmma_f32_16x16x32_bf16(
            false, afr, false, bfr[tn], (short)0, acc[tm][tn], false, false);
    }
#if HAVE_ASYNC_LDS
    wait_async();
#endif
    __syncthreads();
  }

  const i64 co = (i64)z * cBatch;
  #pragma unroll
  for (int tm = 0; tm < 4; ++tm) {
    #pragma unroll
    for (int tn = 0; tn < 2; ++tn) {
      int nb = n0 + wn + tn * 16 + lrow;
      if (nb >= N) continue;
      int mb = m0 + wm + tm * 16 + 8 * half;
      #pragma unroll
      for (int r = 0; r < 8; ++r) {
        int m = mb + r;
        if (m < M) {
          float v = acc[tm][tn][r];
          if (OUT_BF16) ((u16*)Cv)[co + (i64)m * ldc + nb] = f2bf(v);
          else          ((float*)Cv)[co + (i64)m * ldc + nb] = v;
        }
      }
    }
  }
}

// ---------------------------------------------------------------------------
// Flash attention over absorbed latent:  per head, Q'=[q_abs|q_pe] (576) vs
// K'=[c_kv|k_pe] (576); PV uses c_kv (512) via its transpose ckvT.
// Block: 64 queries x 64 keys, 8 waves, causal, online softmax.
// ---------------------------------------------------------------------------
#define FBM 64
#define FBN 64
#define SS_LD 68
#define PS_LD 72

__global__ __launch_bounds__(256)
void fa_kernel(const u16* __restrict__ qabs,  // [H][S][512]
               const u16* __restrict__ qpe,   // [S][H][64]
               const u16* __restrict__ kbuf,  // [S][576]
               const u16* __restrict__ ckvT,  // [512][S]
               u16* __restrict__ obuf,        // [S][H][512]
               float scale)
{
  __shared__ float Ss[FBM * SS_LD];
  __shared__ u16   Ps[FBM * PS_LD];
  __shared__ float alphaS[FBM];
  __shared__ float lS[FBM];

  const int h  = blockIdx.y;
  const int s0 = blockIdx.x * FBM;
  const int tid  = threadIdx.x;
  const int wave = tid >> 5, lane = tid & 31;
  const int half = lane >> 4, lrow = lane & 15;
  const u16* qa = qabs + (i64)h * S_LEN * KVLORA;

  float m_i = -3.0e38f, l_i = 0.f;   // valid for tid < 64

  f32x8 oacc[4][4] = {};             // [m-tile][n-tile]; wave owns 64 V cols
  const int c0w = wave * 64;

  const int nsteps = blockIdx.x + 1;
  for (int step = 0; step < nsteps; ++step) {
    const int t0 = step * FBN;

    // prefetch next KV tile (global_prefetch_b8 hints into GL2/WGP$)
    if (step + 1 < nsteps) {
      const u16* pk = kbuf + (i64)(t0 + FBN + (tid >> 2)) * KDIM + (tid & 3) * 16;
      __builtin_prefetch(pk, 0, 1);
      const u16* pv = ckvT + (i64)(c0w + (lane >> 1)) * S_LEN + t0 + FBN + (lane & 1) * 32;
      __builtin_prefetch(pv, 0, 1);
    }

    // ---- phase 1: scores (each wave: 2 of the 16 16x16 tiles) ----
    #pragma unroll
    for (int u = 0; u < 2; ++u) {
      int idx = wave * 2 + u;
      int tm = idx >> 2, tn = idx & 3;
      int qrow = s0 + tm * 16 + lrow;
      int krow = t0 + tn * 16 + lrow;
      const u16* qaRow = qa + (i64)qrow * KVLORA;
      const u16* qpRow = qpe + ((i64)qrow * NH + h) * DROPE;
      const u16* kRow  = kbuf + (i64)krow * KDIM;
      f32x8 sacc = {};
      #pragma unroll
      for (int kc = 0; kc < 18; ++kc) {
        int f = kc * 32;
        bf16x16 afr = (f < KVLORA) ? load_frag(qaRow + f, half)
                                   : load_frag(qpRow + (f - KVLORA), half);
        bf16x16 bfr = load_frag(kRow + f, half);
        sacc = __builtin_amdgcn_wmma_f32_16x16x32_bf16(
            false, afr, false, bfr, (short)0, sacc, false, false);
      }
      int mB = tm * 16 + 8 * half;
      int nB = tn * 16 + lrow;
      #pragma unroll
      for (int r = 0; r < 8; ++r) {
        int m = mB + r;
        float v = sacc[r] * scale;
        if (t0 + nB > s0 + m) v = -3.0e38f;   // causal mask
        Ss[m * SS_LD + nB] = v;
      }
    }
    __syncthreads();

    // ---- phase 2: online softmax, one thread per query row ----
    if (tid < FBM) {
      const int m = tid;
      float tmax = -3.0e38f;
      for (int j = 0; j < FBN; ++j) tmax = fmaxf(tmax, Ss[m * SS_LD + j]);
      float newm  = fmaxf(m_i, tmax);
      float alpha = __expf(m_i - newm);
      float tsum  = 0.f;
      for (int j = 0; j < FBN; ++j) {
        float pv = __expf(Ss[m * SS_LD + j] - newm);
        tsum += pv;
        Ps[m * PS_LD + j] = f2bf(pv);
      }
      l_i = l_i * alpha + tsum;
      m_i = newm;
      alphaS[m] = alpha;
    }
    __syncthreads();

    // ---- phase 3: O += P @ V  (wave owns V columns [c0w, c0w+64)) ----
    #pragma unroll
    for (int tm = 0; tm < 4; ++tm) {
      float ar[8];
      #pragma unroll
      for (int r = 0; r < 8; ++r) ar[r] = alphaS[tm * 16 + 8 * half + r];
      #pragma unroll
      for (int tn = 0; tn < 4; ++tn)
        #pragma unroll
        for (int r = 0; r < 8; ++r) oacc[tm][tn][r] *= ar[r];
      #pragma unroll
      for (int kc = 0; kc < 2; ++kc) {
        bf16x16 afr = load_frag(&Ps[(tm * 16 + lrow) * PS_LD + kc * 32], half);
        #pragma unroll
        for (int tn = 0; tn < 4; ++tn) {
          int c = c0w + tn * 16 + lrow;
          bf16x16 bfr = load_frag(ckvT + (i64)c * S_LEN + t0 + kc * 32, half);
          oacc[tm][tn] = __builtin_amdgcn_wmma_f32_16x16x32_bf16(
              false, afr, false, bfr, (short)0, oacc[tm][tn], false, false);
        }
      }
    }
    __syncthreads();
  }

  // ---- epilogue: O /= l, store bf16 ----
  if (tid < FBM) lS[tid] = l_i;
  __syncthreads();
  #pragma unroll
  for (int tm = 0; tm < 4; ++tm) {
    float lr[8];
    #pragma unroll
    for (int r = 0; r < 8; ++r) {
      float l = lS[tm * 16 + 8 * half + r];
      lr[r] = (l > 0.f) ? 1.f / l : 0.f;
    }
    #pragma unroll
    for (int tn = 0; tn < 4; ++tn) {
      int c = c0w + tn * 16 + lrow;
      #pragma unroll
      for (int r = 0; r < 8; ++r) {
        int srow = s0 + tm * 16 + 8 * half + r;
        obuf[((i64)srow * NH + h) * KVLORA + c] = f2bf(oacc[tm][tn][r] * lr[r]);
      }
    }
  }
}

// ---------------------------------------------------------------------------
// Elementwise / prep kernels
// ---------------------------------------------------------------------------
__global__ __launch_bounds__(256)
void cast_f32_bf16(const float* __restrict__ in, u16* __restrict__ out, i64 n) {
  i64 stride = (i64)gridDim.x * blockDim.x;
  for (i64 i = (i64)blockIdx.x * blockDim.x + threadIdx.x; i < n; i += stride)
    out[i] = f2bf(in[i]);
}

// w_ukT[h][c][d] = wkv_b[(h*256+d)*512 + c]
__global__ __launch_bounds__(256)
void make_wukT(const float* __restrict__ wkvb, u16* __restrict__ out) {
  const i64 n = (i64)NH * KVLORA * DNOPE;
  i64 stride = (i64)gridDim.x * blockDim.x;
  for (i64 i = (i64)blockIdx.x * blockDim.x + threadIdx.x; i < n; i += stride) {
    int hh = (int)(i >> 16);
    int rr = (int)(i & 65535);
    int c = rr >> 7, d = rr & 127;
    out[i] = f2bf(wkvb[((i64)(hh * 256 + d)) * KVLORA + c]);
  }
}

// w_uv[h][d][c] = wkv_b[(h*256+128+d)*512 + c]
__global__ __launch_bounds__(256)
void make_wuv(const float* __restrict__ wkvb, u16* __restrict__ out) {
  const i64 n = (i64)NH * DV * KVLORA;
  i64 stride = (i64)gridDim.x * blockDim.x;
  for (i64 i = (i64)blockIdx.x * blockDim.x + threadIdx.x; i < n; i += stride) {
    int hh = (int)(i >> 16);
    int rr = (int)(i & 65535);
    int d = rr >> 9, c = rr & 511;
    out[i] = f2bf(wkvb[((i64)(hh * 256 + 128 + d)) * KVLORA + c]);
  }
}

// ckvT[c][t] = kbuf[t*576 + c]   (c < 512)
__global__ __launch_bounds__(256)
void make_ckvT(const u16* __restrict__ kb, u16* __restrict__ out) {
  const i64 n = (i64)KVLORA * S_LEN;
  i64 stride = (i64)gridDim.x * blockDim.x;
  for (i64 i = (i64)blockIdx.x * blockDim.x + threadIdx.x; i < n; i += stride) {
    int c = (int)(i >> 12);
    int t = (int)(i & 4095);
    out[i] = kb[(i64)t * KDIM + c];
  }
}

__global__ __launch_bounds__(256)
void rmsnorm_cast(const float* __restrict__ in, const float* __restrict__ w,
                  u16* __restrict__ out, int C) {
  __shared__ float red[256];
  const int row = blockIdx.x, tid = threadIdx.x;
  const float* x = in + (i64)row * C;
  float ss = 0.f;
  for (int c = tid; c < C; c += 256) { float v = x[c]; ss += v * v; }
  red[tid] = ss; __syncthreads();
  for (int o = 128; o > 0; o >>= 1) {
    if (tid < o) red[tid] += red[tid + o];
    __syncthreads();
  }
  float rinv = rsqrtf(red[0] / (float)C + EPS);
  for (int c = tid; c < C; c += 256) out[(i64)row * C + c] = f2bf(x[c] * rinv * w[c]);
}

// kv row -> kbuf = [rmsnorm(c_kv) | rope(k_pe)]  (bf16, 576 wide)
__global__ __launch_bounds__(256)
void kv_prep(const float* __restrict__ kv, const float* __restrict__ w,
             const float* __restrict__ cosb, const float* __restrict__ sinb,
             u16* __restrict__ kb) {
  __shared__ float red[256];
  const int s = blockIdx.x, tid = threadIdx.x;
  const float* x = kv + (i64)s * KDIM;
  float ss = 0.f;
  for (int c = tid; c < KVLORA; c += 256) { float v = x[c]; ss += v * v; }
  red[tid] = ss; __syncthreads();
  for (int o = 128; o > 0; o >>= 1) {
    if (tid < o) red[tid] += red[tid + o];
    __syncthreads();
  }
  float rinv = rsqrtf(red[0] / (float)KVLORA + EPS);
  for (int c = tid; c < KVLORA; c += 256) kb[(i64)s * KDIM + c] = f2bf(x[c] * rinv * w[c]);
  if (tid < 32) {
    int j = tid;
    float xr = x[KVLORA + 2 * j], xi = x[KVLORA + 2 * j + 1];
    float cs = cosb[s * 32 + j], sn = sinb[s * 32 + j];
    kb[(i64)s * KDIM + KVLORA + 2 * j]     = f2bf(xr * cs - xi * sn);
    kb[(i64)s * KDIM + KVLORA + 2 * j + 1] = f2bf(xr * sn + xi * cs);
  }
}

// q row -> bf16 copy (for absorb GEMM) + roped q_pe
__global__ __launch_bounds__(256)
void q_prep(const float* __restrict__ q, const float* __restrict__ cosb,
            const float* __restrict__ sinb, u16* __restrict__ qb,
            u16* __restrict__ qpe) {
  const int s = blockIdx.x, tid = threadIdx.x;
  const float* x = q + (i64)s * (NH * DHEAD);
  for (int c = tid; c < NH * DHEAD; c += 256) qb[(i64)s * (NH * DHEAD) + c] = f2bf(x[c]);
  for (int p = tid; p < NH * 32; p += 256) {
    int hh = p >> 5, j = p & 31;
    float xr = x[hh * DHEAD + DNOPE + 2 * j], xi = x[hh * DHEAD + DNOPE + 2 * j + 1];
    float cs = cosb[s * 32 + j], sn = sinb[s * 32 + j];
    i64 o = ((i64)s * NH + hh) * DROPE;
    qpe[o + 2 * j]     = f2bf(xr * cs - xi * sn);
    qpe[o + 2 * j + 1] = f2bf(xr * sn + xi * cs);
  }
}

// ---------------------------------------------------------------------------
extern "C" void kernel_launch(void* const* d_in, const int* in_sizes, int n_in,
                              void* d_out, int out_size, void* d_ws, size_t ws_size,
                              hipStream_t stream) {
  const float* x         = (const float*)d_in[0];
  const float* fcos      = (const float*)d_in[2];
  const float* fsin      = (const float*)d_in[3];
  const float* wq_a      = (const float*)d_in[4];
  const float* q_norm_w  = (const float*)d_in[5];
  const float* wq_b      = (const float*)d_in[6];
  const float* wkv_a     = (const float*)d_in[7];
  const float* kv_norm_w = (const float*)d_in[8];
  const float* wkv_b     = (const float*)d_in[9];
  const float* wo        = (const float*)d_in[10];
  float* out = (float*)d_out;

  char* p = (char*)d_ws;
  auto carve = [&](size_t bytes) -> char* {
    char* r = p;
    p += (bytes + 255) & ~(size_t)255;
    return r;
  };
  u16*   x_bf    = (u16*)carve((size_t)S_LEN * DIM * 2);
  u16*   wqa_bf  = (u16*)carve((size_t)QLORA * DIM * 2);
  u16*   wqb_bf  = (u16*)carve((size_t)NH * DHEAD * QLORA * 2);
  u16*   wkva_bf = (u16*)carve((size_t)KDIM * DIM * 2);
  u16*   wukT_bf = (u16*)carve((size_t)NH * KVLORA * DNOPE * 2);
  u16*   wuv_bf  = (u16*)carve((size_t)NH * DV * KVLORA * 2);
  u16*   wo_bf   = (u16*)carve((size_t)DIM * DIM * 2);
  float* qa_f    = (float*)carve((size_t)S_LEN * QLORA * 4);
  u16*   qan_bf  = (u16*)carve((size_t)S_LEN * QLORA * 2);
  float* q_f     = (float*)carve((size_t)S_LEN * NH * DHEAD * 4);
  u16*   qb_bf   = (u16*)carve((size_t)S_LEN * NH * DHEAD * 2);
  u16*   qpe_bf  = (u16*)carve((size_t)S_LEN * NH * DROPE * 2);
  float* kv_f    = (float*)carve((size_t)S_LEN * KDIM * 4);
  u16*   kb_bf   = (u16*)carve((size_t)S_LEN * KDIM * 2);
  u16*   ckvT_bf = (u16*)carve((size_t)KVLORA * S_LEN * 2);
  u16*   qabs_bf = (u16*)carve((size_t)NH * S_LEN * KVLORA * 2);
  u16*   o_bf    = (u16*)carve((size_t)S_LEN * NH * KVLORA * 2);
  u16*   o2_bf   = (u16*)carve((size_t)S_LEN * DIM * 2);

  dim3 blk(256);

  // weight/activation casts + layout transforms
  cast_f32_bf16<<<2048, blk, 0, stream>>>(x,     x_bf,    (i64)S_LEN * DIM);
  cast_f32_bf16<<<1024, blk, 0, stream>>>(wq_a,  wqa_bf,  (i64)QLORA * DIM);
  cast_f32_bf16<<<1024, blk, 0, stream>>>(wq_b,  wqb_bf,  (i64)NH * DHEAD * QLORA);
  cast_f32_bf16<<<1024, blk, 0, stream>>>(wkv_a, wkva_bf, (i64)KDIM * DIM);
  cast_f32_bf16<<<1024, blk, 0, stream>>>(wo,    wo_bf,   (i64)DIM * DIM);
  make_wukT<<<1024, blk, 0, stream>>>(wkv_b, wukT_bf);
  make_wuv <<<1024, blk, 0, stream>>>(wkv_b, wuv_bf);

  // q_a = x @ wq_a^T ; rmsnorm ; q = qan @ wq_b^T
  gemm_bf16<false><<<dim3(8, 32, 1),  blk, 0, stream>>>(x_bf, wqa_bf, qa_f,
      S_LEN, QLORA, DIM, DIM, DIM, QLORA, 0, 0, 0);
  rmsnorm_cast<<<S_LEN, blk, 0, stream>>>(qa_f, q_norm_w, qan_bf, QLORA);
  gemm_bf16<false><<<dim3(24, 32, 1), blk, 0, stream>>>(qan_bf, wqb_bf, q_f,
      S_LEN, NH * DHEAD, QLORA, QLORA, QLORA, NH * DHEAD, 0, 0, 0);

  // kv = x @ wkv_a^T ; build K' = [rmsnorm(c_kv) | rope(k_pe)] ; transpose c_kv
  gemm_bf16<false><<<dim3(5, 32, 1),  blk, 0, stream>>>(x_bf, wkva_bf, kv_f,
      S_LEN, KDIM, DIM, DIM, DIM, KDIM, 0, 0, 0);
  kv_prep<<<S_LEN, blk, 0, stream>>>(kv_f, kv_norm_w, fcos, fsin, kb_bf);
  make_ckvT<<<2048, blk, 0, stream>>>(kb_bf, ckvT_bf);

  // q prep (bf16 copy + rope) ; absorbed q_abs = q_nope @ w_uk  (per head)
  q_prep<<<S_LEN, blk, 0, stream>>>(q_f, fcos, fsin, qb_bf, qpe_bf);
  gemm_bf16<true><<<dim3(4, 32, NH), blk, 0, stream>>>(qb_bf, wukT_bf, qabs_bf,
      S_LEN, KVLORA, DNOPE, NH * DHEAD, DNOPE, KVLORA,
      (i64)DHEAD, (i64)KVLORA * DNOPE, (i64)S_LEN * KVLORA);

  // flash attention in latent space
  fa_kernel<<<dim3(S_LEN / FBM, NH), blk, 0, stream>>>(
      qabs_bf, qpe_bf, kb_bf, ckvT_bf, o_bf, FA_SCALE);

  // per-head output projection: o2[:, h*128:+128] = o_h @ w_uv_h^T
  gemm_bf16<true><<<dim3(1, 32, NH), blk, 0, stream>>>(o_bf, wuv_bf, o2_bf,
      S_LEN, DV, KVLORA, NH * KVLORA, KVLORA, DIM,
      (i64)KVLORA, (i64)DV * KVLORA, (i64)DV);

  // final: out = o2 @ wo^T  (fp32 output)
  gemm_bf16<false><<<dim3(16, 32, 1), blk, 0, stream>>>(o2_bf, wo_bf, out,
      S_LEN, DIM, DIM, DIM, DIM, DIM, 0, 0, 0);
}
